// CrossAttentionFast_29205777613321
// MI455X (gfx1250) — matrix operand order repaired and checked by reading, not compiled, not run on hardware
//
#include <hip/hip_runtime.h>
#include <hip/hip_bf16.h>

// ---------------------------------------------------------------------------
// CDNA5 / gfx1250 WMMA implementation of SR cross-attention.
// B=4, C=320, W=H=64 -> N=4096, SR=2 -> Nk=1024, heads=5, hd=64.
// All GEMMs in bf16 via v_wmma_f32_16x16x32_bf16 (f32 accumulate).
// Cross-lane softmax reductions via DPP8/DPP16 (no LDS in the hot path).
// ---------------------------------------------------------------------------

typedef __bf16 bf16;
typedef __attribute__((ext_vector_type(16))) __bf16 v16bf;
typedef __attribute__((ext_vector_type(8)))  __bf16 v8bf;
typedef __attribute__((ext_vector_type(4)))  __bf16 v4bf;
typedef __attribute__((ext_vector_type(8)))  float  v8f;

#define WMMA_BF16(a, b, c) \
  __builtin_amdgcn_wmma_f32_16x16x32_bf16(false, (a), false, (b), (short)0, (c), false, false)

// --- fragment loaders (per CDNA5 ISA 7.12.2 VGPR layouts, wave32) -----------
// A-matrix 16x32 bf16: lane r=lane&15 -> row M=r; half=lane>>4:
//   elements 0..7  = K = half*8 + 0..7       (VGPR 0..3)
//   elements 8..15 = K = 16 + half*8 + 0..7  (VGPR 4..7)
__device__ __forceinline__ v16bf load_afrag(const bf16* base, int ld, int row0,
                                            int k0, int lane) {
  const int half = lane >> 4, r = lane & 15;
  const bf16* p = base + (size_t)(row0 + r) * ld + (k0 + half * 8);
  v8bf lo = *(const v8bf*)p;
  v8bf hi = *(const v8bf*)(p + 16);
  return __builtin_shufflevector(lo, hi, 0, 1, 2, 3, 4, 5, 6, 7,
                                 8, 9, 10, 11, 12, 13, 14, 15);
}

// B-matrix 32x16 bf16: lane r -> column N=r; half selects K block:
//   elements 0..7 = K = half*8 + 0..7, elements 8..15 = K = 16 + half*8 + 0..7
// Column N of B comes from row N of W when computing A*W^T (K-contiguous).
__device__ __forceinline__ v16bf load_bfrag(const bf16* base, int ld, int col0,
                                            int k0, int lane) {
  const int half = lane >> 4, r = lane & 15;
  const bf16* p = base + (size_t)(col0 + r) * ld + (k0 + half * 8);
  v8bf lo = *(const v8bf*)p;
  v8bf hi = *(const v8bf*)(p + 16);
  return __builtin_shufflevector(lo, hi, 0, 1, 2, 3, 4, 5, 6, 7,
                                 8, 9, 10, 11, 12, 13, 14, 15);
}

// --- VALU-only cross-lane xor exchanges (groups of 16) ----------------------
// DPP8 handles xor 1/2/4 inside groups of 8; DPP16 ROW_MIRROR is lane^15.
// {1,2,4,15} generates Z2^4, so the 4-step butterfly reduces all 16 lanes.
template <int OFF>
__device__ __forceinline__ float dpp8_xor(float v) {
  constexpr int sel = ((0 ^ OFF) << 0) | ((1 ^ OFF) << 3) | ((2 ^ OFF) << 6) |
                      ((3 ^ OFF) << 9) | ((4 ^ OFF) << 12) |
                      ((5 ^ OFF) << 15) | ((6 ^ OFF) << 18) |
                      ((7 ^ OFF) << 21);
  return __int_as_float(__builtin_amdgcn_mov_dpp8(__float_as_int(v), sel));
}
__device__ __forceinline__ float dpp16_mirror(float v) {
  // DPP_CTRL 0x140 = ROW_MIRROR (lane ^ 15 within each row of 16)
  return __int_as_float(__builtin_amdgcn_update_dpp(
      0, __float_as_int(v), 0x140, 0xf, 0xf, true));
}
__device__ __forceinline__ float red_max16(float v) {
  v = fmaxf(v, dpp8_xor<1>(v));
  v = fmaxf(v, dpp8_xor<2>(v));
  v = fmaxf(v, dpp8_xor<4>(v));
  v = fmaxf(v, dpp16_mirror(v));
  return v;
}
__device__ __forceinline__ float red_sum16(float v) {
  v += dpp8_xor<1>(v);
  v += dpp8_xor<2>(v);
  v += dpp8_xor<4>(v);
  v += dpp16_mirror(v);
  return v;
}

// --- elementwise f32 -> bf16 convert ---------------------------------------
__global__ void cvt_bf16(const float* __restrict__ in, bf16* __restrict__ out,
                         int n) {
  int i = blockIdx.x * 256 + threadIdx.x;
  if (i < n) out[i] = (bf16)in[i];
}

// --- query (B,320,4096) -> qf (B,4096,320) bf16, LDS-tiled transpose -------
__global__ __launch_bounds__(256) void transpose_cvt(
    const float* __restrict__ in, bf16* __restrict__ out) {
  __shared__ float tile[32][33];
  const int tx = threadIdx.x, ty = threadIdx.y;  // 32 x 8
  const int n0 = blockIdx.x * 32, c0 = blockIdx.y * 32, b = blockIdx.z;
#pragma unroll
  for (int i = 0; i < 4; ++i)
    tile[ty + 8 * i][tx] =
        in[(size_t)(b * 320 + c0 + ty + 8 * i) * 4096 + n0 + tx];
  __syncthreads();
#pragma unroll
  for (int i = 0; i < 4; ++i)
    out[(size_t)(b * 4096 + n0 + ty + 8 * i) * 320 + c0 + tx] =
        (bf16)tile[tx][ty + 8 * i];
}

// --- im2row for 2x2 stride-2 conv: patch[(b*1024+m), ci*4 + ki*2 + kj] ------
__global__ void im2row_conv(const float* __restrict__ x,
                            bf16* __restrict__ patch) {
  int idx = blockIdx.x * 256 + threadIdx.x;  // (b*1024+m)*320 + ci
  if (idx >= 4 * 1024 * 320) return;
  const int ci = idx % 320;
  const int bm = idx / 320;
  const int m = bm & 1023;
  const int b = bm >> 10;
  const int oi = m >> 5, oj = m & 31;
  const float* s = x + (((size_t)(b * 320 + ci) * 64 + 2 * oi) * 64 + 2 * oj);
  v4bf w;
  w[0] = (bf16)s[0];
  w[1] = (bf16)s[1];
  w[2] = (bf16)s[64];
  w[3] = (bf16)s[65];
  *(v4bf*)(patch + (size_t)bm * 1280 + ci * 4) = w;
}

// --- generic WMMA GEMM: Y(MxN) = A(MxK) * W(NxK)^T, one wave = 32x64 tile ---
// MODE 0: bf16 row-major, no bias.
// MODE 1: f32 row-major + bias.
// MODE 2: f32 scattered to NCHW output (row = b*4096+n) + bias.
template <int MODE>
__global__ __launch_bounds__(32) void gemm_wmma(
    const bf16* __restrict__ A, const bf16* __restrict__ W,
    const float* __restrict__ bias, void* __restrict__ out, int M, int N,
    int K) {
  const int lane = threadIdx.x;
  const int m0 = blockIdx.x * 32;
  const int n0 = blockIdx.y * 64;
  v8f acc[2][4] = {};
  for (int k = 0; k < K; k += 32) {
    v16bf a0 = load_afrag(A, K, m0, k, lane);
    v16bf a1 = load_afrag(A, K, m0 + 16, k, lane);
#pragma unroll
    for (int ni = 0; ni < 4; ++ni) {
      v16bf bfrag = load_bfrag(W, K, n0 + ni * 16, k, lane);
      acc[0][ni] = WMMA_BF16(a0, bfrag, acc[0][ni]);
      acc[1][ni] = WMMA_BF16(a1, bfrag, acc[1][ni]);
    }
  }
  const int half = lane >> 4, r = lane & 15;
#pragma unroll
  for (int mi = 0; mi < 2; ++mi)
#pragma unroll
    for (int ni = 0; ni < 4; ++ni) {
      const int col = n0 + ni * 16 + r;
#pragma unroll
      for (int j = 0; j < 8; ++j) {
        const int row = m0 + mi * 16 + j + 8 * half;
        const float v = acc[mi][ni][j];
        if (MODE == 0) {
          ((bf16*)out)[(size_t)row * N + col] = (bf16)v;
        } else if (MODE == 1) {
          ((float*)out)[(size_t)row * N + col] = v + bias[col];
        } else {
          const int b = row >> 12, n = row & 4095;
          ((float*)out)[((size_t)b * 320 + col) * 4096 + n] = v + bias[col];
        }
      }
    }
}

// --- LayerNorm over C=320, f32 in -> bf16 out -------------------------------
__global__ __launch_bounds__(64) void layernorm_bf16(
    const float* __restrict__ in, const float* __restrict__ g,
    const float* __restrict__ bta, bf16* __restrict__ out) {
  const int row = blockIdx.x;  // 0..4095 (b*1024+m)
  const int t = threadIdx.x;   // 0..63
  const float* p = in + (size_t)row * 320;
  float v[5], s = 0.f, ss = 0.f;
#pragma unroll
  for (int i = 0; i < 5; ++i) {
    v[i] = p[t + 64 * i];
    s += v[i];
    ss += v[i] * v[i];
  }
#pragma unroll
  for (int off = 1; off < 32; off <<= 1) {
    s += __shfl_xor(s, off, 32);
    ss += __shfl_xor(ss, off, 32);
  }
  __shared__ float wsum[2], wsq[2];
  if ((t & 31) == 0) {
    wsum[t >> 5] = s;
    wsq[t >> 5] = ss;
  }
  __syncthreads();
  s = wsum[0] + wsum[1];
  ss = wsq[0] + wsq[1];
  const float mu = s * (1.f / 320.f);
  const float var = ss * (1.f / 320.f) - mu * mu;
  const float rstd = rsqrtf(var + 1e-5f);
  bf16* o = out + (size_t)row * 320;
#pragma unroll
  for (int i = 0; i < 5; ++i) {
    const int c = t + 64 * i;
    o[c] = (bf16)((v[i] - mu) * rstd * g[c] + bta[c]);
  }
}

// --- repack q row-major (b*4096+n, h*64+d) -> (b,h,n,d) ---------------------
__global__ void repack_q(const bf16* __restrict__ in, bf16* __restrict__ out) {
  int idx = blockIdx.x * 256 + threadIdx.x;
  if (idx >= 16384 * 320) return;
  const int row = idx / 320, col = idx % 320;
  const int b = row >> 12, n = row & 4095;
  const int h = col >> 6, d = col & 63;
  out[((size_t)(b * 5 + h) * 4096 + n) * 64 + d] = in[idx];
}

// --- repack kv (b*1024+m, s*320+h*64+d) -> k(b,h,m,d), vT(b,h,d,m) ----------
__global__ void repack_kv(const bf16* __restrict__ in, bf16* __restrict__ kout,
                          bf16* __restrict__ vout) {
  int idx = blockIdx.x * 256 + threadIdx.x;
  if (idx >= 4096 * 640) return;
  const int row = idx / 640, col = idx % 640;
  const int b = row >> 10, m = row & 1023;
  const int c = col % 320;
  const int h = c >> 6, d = c & 63;
  const bf16 val = in[idx];
  if (col < 320)
    kout[((size_t)(b * 5 + h) * 1024 + m) * 64 + d] = val;
  else
    vout[((size_t)(b * 5 + h) * 64 + d) * 1024 + m] = val;
}

// --- fused flash attention: softmax(q k^T * 0.125) v ------------------------
// grid: (N/64, heads, B); block = 128 (4 waves); each wave owns 16 query rows.
__global__ __launch_bounds__(128) void attn_flash(
    const bf16* __restrict__ q, const bf16* __restrict__ k,
    const bf16* __restrict__ vT, bf16* __restrict__ out) {
  __shared__ alignas(16) bf16 pbuf[4][16 * 32];
  const int lane = threadIdx.x & 31;
  const int wave = threadIdx.x >> 5;
  const int h = blockIdx.y, b = blockIdx.z;
  const int n0 = blockIdx.x * 64 + wave * 16;
  const size_t bh = (size_t)(b * 5 + h);
  const bf16* qb = q + bh * 4096 * 64;   // (n, d)
  const bf16* kb = k + bh * 1024 * 64;   // (m, d)
  const bf16* vb = vT + bh * 64 * 1024;  // (d, m)

  const v16bf qA0 = load_afrag(qb, 64, n0, 0, lane);   // K(d) 0..31
  const v16bf qA1 = load_afrag(qb, 64, n0, 32, lane);  // K(d) 32..63

  const int half = lane >> 4, r = lane & 15;
  float mrow[8], lrow[8];
#pragma unroll
  for (int j = 0; j < 8; ++j) {
    mrow[j] = -1e30f;
    lrow[j] = 0.f;
  }
  v8f acc[4] = {};
  const float scale = 0.125f;  // 64^-0.5
  bf16* pb = &pbuf[wave][0];

  for (int mt = 0; mt < 1024; mt += 32) {
    // S = q * k^T for columns [mt, mt+32)
    v16bf kB0 = load_bfrag(kb, 64, mt, 0, lane);
    v16bf kB1 = load_bfrag(kb, 64, mt, 32, lane);
    v16bf kB2 = load_bfrag(kb, 64, mt + 16, 0, lane);
    v16bf kB3 = load_bfrag(kb, 64, mt + 16, 32, lane);
    v8f s0 = {}, s1 = {};
    s0 = WMMA_BF16(qA0, kB0, s0);
    s0 = WMMA_BF16(qA1, kB1, s0);
    s1 = WMMA_BF16(qA0, kB2, s1);
    s1 = WMMA_BF16(qA1, kB3, s1);

    // scale + per-row chunk max (VALU-only butterfly across the 16 col-lanes)
    float cmax[8];
#pragma unroll
    for (int j = 0; j < 8; ++j) {
      s0[j] *= scale;
      s1[j] *= scale;
      cmax[j] = red_max16(fmaxf(s0[j], s1[j]));
    }

    // online softmax update; P -> LDS in A-fragment-friendly row-major
#pragma unroll
    for (int j = 0; j < 8; ++j) {
      const float mnew = fmaxf(mrow[j], cmax[j]);
      const float alpha = __expf(mrow[j] - mnew);
      mrow[j] = mnew;
      const float p0 = __expf(s0[j] - mnew);
      const float p1 = __expf(s1[j] - mnew);
      lrow[j] = lrow[j] * alpha + p0 + p1;
      acc[0][j] *= alpha;
      acc[1][j] *= alpha;
      acc[2][j] *= alpha;
      acc[3][j] *= alpha;
      const int row = j + 8 * half;
      pb[row * 32 + r] = (bf16)p0;
      pb[row * 32 + 16 + r] = (bf16)p1;
    }
    asm volatile("s_wait_dscnt 0" ::: "memory");
    const v16bf pA = load_afrag(pb, 32, 0, 0, lane);  // 16x32 of P
#pragma unroll
    for (int f = 0; f < 4; ++f) {
      v16bf vB = load_bfrag(vb, 1024, f * 16, mt, lane);
      acc[f] = WMMA_BF16(pA, vB, acc[f]);
    }
  }

  // final 1/l, write (B,N,C) bf16
#pragma unroll
  for (int j = 0; j < 8; ++j) lrow[j] = 1.f / red_sum16(lrow[j]);

#pragma unroll
  for (int f = 0; f < 4; ++f)
#pragma unroll
    for (int j = 0; j < 8; ++j) {
      const int n = n0 + j + 8 * half;
      const int c = h * 64 + f * 16 + r;
      out[((size_t)b * 4096 + n) * 320 + c] = (bf16)(acc[f][j] * lrow[j]);
    }
}

// ---------------------------------------------------------------------------
extern "C" void kernel_launch(void* const* d_in, const int* in_sizes, int n_in,
                              void* d_out, int out_size, void* d_ws,
                              size_t ws_size, hipStream_t stream) {
  const float* query = (const float*)d_in[0];
  const float* x = (const float*)d_in[1];
  const float* q_w = (const float*)d_in[2];
  const float* kv_w = (const float*)d_in[3];
  const float* proj_w = (const float*)d_in[4];
  const float* proj_b = (const float*)d_in[5];
  const float* sr_w = (const float*)d_in[6];
  const float* sr_b = (const float*)d_in[7];
  const float* ln_g = (const float*)d_in[8];
  const float* ln_b = (const float*)d_in[9];
  float* out = (float*)d_out;
  (void)in_sizes;
  (void)n_in;
  (void)out_size;
  (void)ws_size;

  // workspace layout (256B aligned)
  char* ws = (char*)d_ws;
  size_t off = 0;
  auto take = [&](size_t bytes) {
    char* p = ws + off;
    off += (bytes + 255) & ~(size_t)255;
    return p;
  };
  bf16* wq = (bf16*)take(320 * 320 * 2);
  bf16* wkv = (bf16*)take(640 * 320 * 2);
  bf16* wpr = (bf16*)take(320 * 320 * 2);
  bf16* wsr = (bf16*)take((size_t)320 * 1280 * 2);
  bf16* qf = (bf16*)take((size_t)16384 * 320 * 2);
  bf16* qY = (bf16*)take((size_t)16384 * 320 * 2);  // reused as attn output
  bf16* qpk = (bf16*)take((size_t)16384 * 320 * 2);
  bf16* patch = (bf16*)take((size_t)4096 * 1280 * 2);
  float* convf = (float*)take((size_t)4096 * 320 * 4);
  bf16* xln = (bf16*)take((size_t)4096 * 320 * 2);
  bf16* kvY = (bf16*)take((size_t)4096 * 640 * 2);
  bf16* kpk = (bf16*)take((size_t)4 * 5 * 1024 * 64 * 2);
  bf16* vT = (bf16*)take((size_t)4 * 5 * 64 * 1024 * 2);
  bf16* attnout = qY;

  // 1) convert weights to bf16
  cvt_bf16<<<(320 * 320 + 255) / 256, 256, 0, stream>>>(q_w, wq, 320 * 320);
  cvt_bf16<<<(640 * 320 + 255) / 256, 256, 0, stream>>>(kv_w, wkv, 640 * 320);
  cvt_bf16<<<(320 * 320 + 255) / 256, 256, 0, stream>>>(proj_w, wpr, 320 * 320);
  cvt_bf16<<<(320 * 1280 + 255) / 256, 256, 0, stream>>>(sr_w, wsr, 320 * 1280);

  // 2) query NCHW -> (B,N,C) bf16
  transpose_cvt<<<dim3(128, 10, 4), dim3(32, 8, 1), 0, stream>>>(query, qf);

  // 3) q = qf @ q_w^T  (M=16384, N=320, K=320), then pack (b,h,n,d)
  gemm_wmma<0><<<dim3(512, 5), 32, 0, stream>>>(qf, wq, nullptr, qY, 16384,
                                                320, 320);
  repack_q<<<(16384 * 320 + 255) / 256, 256, 0, stream>>>(qY, qpk);

  // 4) spatial reduction conv as GEMM: im2row then (M=4096,N=320,K=1280)+bias
  im2row_conv<<<(4 * 1024 * 320 + 255) / 256, 256, 0, stream>>>(x, patch);
  gemm_wmma<1><<<dim3(128, 5), 32, 0, stream>>>(patch, wsr, sr_b, convf, 4096,
                                                320, 1280);

  // 5) LayerNorm over C -> bf16
  layernorm_bf16<<<4096, 64, 0, stream>>>(convf, ln_g, ln_b, xln);

  // 6) kv = xln @ kv_w^T (M=4096, N=640, K=320); split to k and v^T
  gemm_wmma<0><<<dim3(128, 10), 32, 0, stream>>>(xln, wkv, nullptr, kvY, 4096,
                                                 640, 320);
  repack_kv<<<(4096 * 640 + 255) / 256, 256, 0, stream>>>(kvY, kpk, vT);

  // 7) fused flash attention -> (B,N,C) bf16
  attn_flash<<<dim3(64, 5, 4), 128, 0, stream>>>(qpk, kpk, vT, attnout);

  // 8) out projection + bias, scattered straight to NCHW f32 output
  gemm_wmma<2><<<dim3(512, 5), 32, 0, stream>>>(attnout, wpr, proj_b, out,
                                                16384, 320, 320);
}
